// Net_8366596293162
// MI455X (gfx1250) — matrix operand order repaired and checked by reading, not compiled, not run on hardware
//
#include <hip/hip_runtime.h>
#include <math.h>

// ---------------------------------------------------------------------------
// Problem constants (from reference)
// ---------------------------------------------------------------------------
#define N_NODES 50000
#define N_EDGES 800000

typedef _Float16 v16h __attribute__((ext_vector_type(16)));
typedef float    v8f  __attribute__((ext_vector_type(8)));

// ---------------------------------------------------------------------------
// WMMA GEMM:  Y[N x Nout] = op( X[N x K] @ W[K x Nout] + bias ) (+ bias2)
//   mode 0: Y = X@W (+bias)
//   mode 1: Y = relu(X@W + bias)
//   mode 2: Y = relu(X@W + bias) + bias2
// X may be a concatenation of X1 (first K1 cols) and X2 (rest).
// W is fp16 pre-swizzled into WMMA-B fragment order (see pack_w):
//   fragment f = kt*NT + nt is 512 contiguous halves; lane l owns halves
//   [f*512 + l*16, +16) == one v16h == two global_load_b128 per fragment.
// One block = 128 threads = 4 waves; each wave computes a 16-row strip.
// ---------------------------------------------------------------------------
#define GEMM_ROWS 64
#define LDS_ELEMS (64 * 264)   // 64 rows x (256 + 8 skew) halves = 33 KB

template <int NT>
__global__ __launch_bounds__(128) void gemm_wmma(
    const float* __restrict__ X1, int ldx1, int K1,
    const float* __restrict__ X2, int ldx2,
    const _Float16* __restrict__ W,
    const float* __restrict__ bias, const float* __restrict__ bias2,
    int nValid, float* __restrict__ Y, int ldy, int nrows, int K, int mode)
{
    __shared__ _Float16 sA[LDS_ELEMS];
    const int tid     = threadIdx.x;
    const int lane    = tid & 31;
    const int wave    = tid >> 5;
    const int rowBase = blockIdx.x * GEMM_ROWS;

    const int Kpad  = (K + 31) & ~31;        // always a power of two here
    const int ldsS  = Kpad + 8;              // skew to avoid bank conflicts
    const int kbits = 31 - __clz(Kpad);

    // Stage fp32 activations -> fp16 LDS tile (zero-padded rows/cols).
    for (int idx = tid; idx < GEMM_ROWS * Kpad; idx += 128) {
        const int r   = idx >> kbits;
        const int k   = idx & (Kpad - 1);
        const int row = rowBase + r;
        float v = 0.0f;
        if (row < nrows && k < K) {
            v = (k < K1) ? X1[(size_t)row * ldx1 + k]
                         : X2[(size_t)row * ldx2 + (k - K1)];
        }
        sA[r * ldsS + k] = (_Float16)v;
    }
    __syncthreads();

    v8f zero = {};
    v8f acc[NT];
#pragma unroll
    for (int i = 0; i < NT; ++i) acc[i] = zero;

    const int mlocal = (wave << 4) + (lane & 15);
    const int kbA    = (lane < 16) ? 0 : 16;   // A layout: lane-half K offset
    const _Float16* __restrict__ Wlane = W + (size_t)lane * 16;
    const int ncol   = lane & 15;

    const int kTiles = Kpad >> 5;
    for (int kt = 0; kt < kTiles; ++kt) {
        const int k0 = kt << 5;
        // Build A fragment (16x32 f16, contiguous k = 16h + s, matching pack_w) from LDS.
        v16h a;
#pragma unroll
        for (int v = 0; v < 8; ++v) {
            const int k = k0 + kbA + 2 * v;
            a[2 * v]     = sA[mlocal * ldsS + k];
            a[2 * v + 1] = sA[mlocal * ldsS + k + 1];
        }
#pragma unroll
        for (int nt = 0; nt < NT; ++nt) {
            // B fragment: one coalesced 32-byte load from packed weights.
            const v16h b = *reinterpret_cast<const v16h*>(
                Wlane + ((size_t)(kt * NT + nt) << 9));
            acc[nt] = __builtin_amdgcn_wmma_f32_16x16x32_f16(
                false, a, false, b, (short)0, acc[nt], false, false);
        }
    }

    // Store C/D fragment: VGPR r holds M=r (lanes 0-15) / M=8+r (lanes 16-31).
    const int rowHalf = (lane < 16) ? 0 : 8;
#pragma unroll
    for (int nt = 0; nt < NT; ++nt) {
        const int n   = (nt << 4) + ncol;
        const float bv  = (bias  && n < nValid) ? bias[n]  : 0.0f;
        const float b2v = (bias2 && n < nValid) ? bias2[n] : 0.0f;
#pragma unroll
        for (int r = 0; r < 8; ++r) {
            const int row = rowBase + (wave << 4) + rowHalf + r;
            if (row < nrows) {
                float v = acc[nt][r] + bv;
                if (mode >= 1) v = fmaxf(v, 0.0f);
                if (mode == 2) v += b2v;
                Y[(size_t)row * ldy + n] = v;
            }
        }
    }
}

// ---------------------------------------------------------------------------
// Small utility kernels
// ---------------------------------------------------------------------------
__global__ void zero_f32(float* p, int n) {
    int i = blockIdx.x * blockDim.x + threadIdx.x;
    if (i < n) p[i] = 0.0f;
}

__global__ void deg_kernel(const int* __restrict__ dst, float* __restrict__ deg, int E) {
    int e = blockIdx.x * blockDim.x + threadIdx.x;
    if (e < E) atomicAdd(&deg[dst[e]], 1.0f);
}

__global__ void dinv_kernel(float* __restrict__ d, int n) {   // in-place deg -> dinv
    int i = blockIdx.x * blockDim.x + threadIdx.x;
    if (i < n) {
        float v = d[i];
        d[i] = (v > 0.0f) ? rsqrtf(fmaxf(v, 1e-12f)) : 0.0f;
    }
}

__global__ void norm_kernel(const int* __restrict__ src, const int* __restrict__ dst,
                            const float* __restrict__ dinv, float* __restrict__ nrm, int E) {
    int e = blockIdx.x * blockDim.x + threadIdx.x;
    if (e < E) nrm[e] = dinv[src[e]] * dinv[dst[e]];
}

// fp32 weights -> fp16 packed WMMA-B fragment layout, zero padded to Kpad x Nout.
// Element gid: f = gid/512 (fragment), lane = (gid%512)/16, j = gid%16.
//   kt = f / (Nout/16), nt = f % (Nout/16)
//   value = W[kt*32 + (lane<16?0:16) + j][nt*16 + (lane&15)]
__global__ void pack_w(const float* __restrict__ s, _Float16* __restrict__ d,
                       int srcR, int srcC, int Kpad, int Nout) {
    const int total = Kpad * Nout;
    int gid = blockIdx.x * blockDim.x + threadIdx.x;
    if (gid >= total) return;
    const int f    = gid >> 9;
    const int rem  = gid & 511;
    const int lane = rem >> 4;
    const int j    = rem & 15;
    const int ntiles = Nout >> 4;
    const int kt = f / ntiles;
    const int nt = f - kt * ntiles;
    const int k = (kt << 5) + ((lane < 16) ? 0 : 16) + j;
    const int n = (nt << 4) + (lane & 15);
    d[gid] = (k < srcR && n < srcC) ? (_Float16)s[k * srcC + n] : (_Float16)0.0f;
}

// segment_sum scatter: Out[dst] += norm * H[src]  (width = 4*w4, row stride)
__global__ void scatter_kernel(const float* __restrict__ H, float* __restrict__ Out,
                               const int* __restrict__ src, const int* __restrict__ dst,
                               const float* __restrict__ nrm, int E, int w4, int stride) {
    int gid = blockIdx.x * blockDim.x + threadIdx.x;
    int e = gid / w4;
    if (e >= E) return;
    int c = (gid - e * w4) * 4;
    int s = src[e], d = dst[e];
    float nm = nrm[e];
    const float4 hv = *reinterpret_cast<const float4*>(H + (size_t)s * stride + c);
    float* op = Out + (size_t)d * stride + c;
    atomicAdd(op + 0, nm * hv.x);
    atomicAdd(op + 1, nm * hv.y);
    atomicAdd(op + 2, nm * hv.z);
    atomicAdd(op + 3, nm * hv.w);
}

// attention score: s3[n*3+k] = sum_j tanh(V[n,j]) * u[j]   (one wave / node)
__global__ void att_reduce(const float* __restrict__ V, const float* __restrict__ u,
                           float* __restrict__ s3, int N, int k) {
    int wid  = (blockIdx.x * blockDim.x + threadIdx.x) >> 5;
    int lane = threadIdx.x & 31;
    if (wid >= N) return;
    float acc = 0.0f;
    for (int j = lane; j < 128; j += 32)
        acc += tanhf(V[(size_t)wid * 128 + j]) * u[j];
    for (int off = 16; off > 0; off >>= 1)
        acc += __shfl_down(acc, off, 32);
    if (lane == 0) s3[wid * 3 + k] = acc;
}

// fused[n,d] = sum_k softmax(s+1e-6)[n,k] * emb_k[n,d]
__global__ void softmax_fuse(const float* __restrict__ e1, const float* __restrict__ e2,
                             const float* __restrict__ e3, const float* __restrict__ s3,
                             float* __restrict__ fused, int N) {
    int gid = blockIdx.x * blockDim.x + threadIdx.x;
    if (gid >= N * 128) return;
    int n = gid >> 7, d = gid & 127;
    float s0 = s3[n * 3 + 0] + 1e-6f;
    float s1 = s3[n * 3 + 1] + 1e-6f;
    float s2 = s3[n * 3 + 2] + 1e-6f;
    float m  = fmaxf(s0, fmaxf(s1, s2));
    float x0 = expf(s0 - m), x1 = expf(s1 - m), x2 = expf(s2 - m);
    float inv = 1.0f / (x0 + x1 + x2);
    size_t idx = (size_t)n * 128 + d;
    fused[idx] = (x0 * e1[idx] + x1 * e2[idx] + x2 * e3[idx]) * inv;
}

__global__ void fillb(float* __restrict__ G, const float* __restrict__ b, int total, int w) {
    int i = blockIdx.x * blockDim.x + threadIdx.x;
    if (i < total) G[i] = b[i % w];
}

__global__ void combine_relu(const float* __restrict__ h0, const float* __restrict__ G,
                             float* __restrict__ h, int total) {
    int i = blockIdx.x * blockDim.x + threadIdx.x;
    if (i < total) h[i] = h0[i] + fmaxf(G[i], 0.0f);
}

// out[n] = h[n]·fc2w + fc2b + cv2b ; t[n] = h[n]·cv2w
__global__ void final_init(const float* __restrict__ h, const float* __restrict__ fc2w,
                           const float* __restrict__ fc2b, const float* __restrict__ cv2w,
                           const float* __restrict__ cv2b, float* __restrict__ out,
                           float* __restrict__ t, int N) {
    int n = blockIdx.x * blockDim.x + threadIdx.x;
    if (n >= N) return;
    const float* hr = h + (size_t)n * 64;
    float o = 0.0f, tt = 0.0f;
#pragma unroll
    for (int j = 0; j < 64; ++j) {
        float hv = hr[j];
        o  += hv * fc2w[j];
        tt += hv * cv2w[j];
    }
    out[n] = o + fc2b[0] + cv2b[0];
    t[n]   = tt;
}

__global__ void scatter1(const float* __restrict__ t, float* __restrict__ out,
                         const int* __restrict__ src, const int* __restrict__ dst,
                         const float* __restrict__ nrm, int E) {
    int e = blockIdx.x * blockDim.x + threadIdx.x;
    if (e < E) atomicAdd(&out[dst[e]], nrm[e] * t[src[e]]);
}

// ---------------------------------------------------------------------------
// Host side
// ---------------------------------------------------------------------------
static void gemm(hipStream_t st, const float* X1, int ldx1, int K1,
                 const float* X2, int ldx2, const _Float16* W,
                 const float* bias, const float* bias2, int nValid,
                 float* Y, int ldy, int K, int Nout, int mode)
{
    dim3 g((N_NODES + GEMM_ROWS - 1) / GEMM_ROWS), b(128);
    switch (Nout >> 4) {
    case 8: gemm_wmma<8><<<g, b, 0, st>>>(X1, ldx1, K1, X2, ldx2, W,
                bias, bias2, nValid, Y, ldy, N_NODES, K, mode); break;
    case 7: gemm_wmma<7><<<g, b, 0, st>>>(X1, ldx1, K1, X2, ldx2, W,
                bias, bias2, nValid, Y, ldy, N_NODES, K, mode); break;
    case 4: gemm_wmma<4><<<g, b, 0, st>>>(X1, ldx1, K1, X2, ldx2, W,
                bias, bias2, nValid, Y, ldy, N_NODES, K, mode); break;
    }
}

extern "C" void kernel_launch(void* const* d_in, const int* in_sizes, int n_in,
                              void* d_out, int out_size, void* d_ws, size_t ws_size,
                              hipStream_t stream)
{
    const int N = N_NODES, E = N_EDGES;
    const float* x    = (const float*)d_in[0];
    const int*   ei   = (const int*)d_in[1];
    const int*   esrc = ei;
    const int*   edst = ei + E;
    const float* e12_fc_w = (const float*)d_in[2],  *e12_fc_b = (const float*)d_in[3];
    const float* e12_cv_w = (const float*)d_in[4],  *e12_cv_b = (const float*)d_in[5];
    const float* e23_fc_w = (const float*)d_in[6],  *e23_fc_b = (const float*)d_in[7];
    const float* e23_cv_w = (const float*)d_in[8],  *e23_cv_b = (const float*)d_in[9];
    const float* mlp_w    = (const float*)d_in[10], *mlp_b    = (const float*)d_in[11];
    const float* fc1_w    = (const float*)d_in[12], *fc1_b    = (const float*)d_in[13];
    const float* att_w    = (const float*)d_in[14], *att_u    = (const float*)d_in[15];
    const float* pr_fc_w  = (const float*)d_in[16], *pr_fc_b  = (const float*)d_in[17];
    const float* pr_cv_w  = (const float*)d_in[18], *pr_cv_b  = (const float*)d_in[19];
    const float* cl_fc1_w = (const float*)d_in[20], *cl_fc1_b = (const float*)d_in[21];
    const float* cl_fc2_w = (const float*)d_in[22], *cl_fc2_b = (const float*)d_in[23];
    const float* cl_cv1_w = (const float*)d_in[24], *cl_cv1_b = (const float*)d_in[25];
    const float* cl_cv2_w = (const float*)d_in[26], *cl_cv2_b = (const float*)d_in[27];
    float* out = (float*)d_out;

    // ---- workspace layout (fp32 region then fp16 packed-weight arena) ----
    float* ws = (float*)d_ws;
    size_t off = 0;
    float* dinv = ws + off; off += N;        // also degree (in-place)
    float* nrm  = ws + off; off += E;
    float* s3   = ws + off; off += (size_t)3 * N;
    float* tbuf = ws + off; off += N;
    float* B[6];
    for (int i = 0; i < 6; ++i) { B[i] = ws + off; off += (size_t)N * 128; }
    _Float16* hw = (_Float16*)(ws + off);
    _Float16* h_e12_fc = hw;                 // Kpad=32  x 128
    _Float16* h_e12_cv = hw + 4096;
    _Float16* h_e23_fc = hw + 8192;
    _Float16* h_e23_cv = hw + 12288;
    _Float16* h_mlp    = hw + 16384;         // 128 x 128
    _Float16* h_fc1    = hw + 32768;         // 256 x 128
    _Float16* h_att    = hw + 65536;         // 128 x 128
    _Float16* h_prfc   = hw + 81920;         // 128 x 112
    _Float16* h_prcv   = hw + 96256;         // 128 x 112
    _Float16* h_clfc1  = hw + 110592;        // 128 x 64
    _Float16* h_clcv1  = hw + 118784;        // 128 x 64

    dim3 b256(256);
    auto g1 = [](int n) { return dim3((n + 255) / 256); };

    // ---- fp16 weight packing into WMMA-B fragment layout ----
    pack_w<<<g1(32*128),  b256, 0, stream>>>(e12_fc_w, h_e12_fc, 16, 128, 32, 128);
    pack_w<<<g1(32*128),  b256, 0, stream>>>(e12_cv_w, h_e12_cv, 16, 128, 32, 128);
    pack_w<<<g1(32*128),  b256, 0, stream>>>(e23_fc_w, h_e23_fc, 16, 128, 32, 128);
    pack_w<<<g1(32*128),  b256, 0, stream>>>(e23_cv_w, h_e23_cv, 16, 128, 32, 128);
    pack_w<<<g1(128*128), b256, 0, stream>>>(mlp_w,    h_mlp,   128, 128, 128, 128);
    pack_w<<<g1(256*128), b256, 0, stream>>>(fc1_w,    h_fc1,   256, 128, 256, 128);
    pack_w<<<g1(128*128), b256, 0, stream>>>(att_w,    h_att,   128, 128, 128, 128);
    pack_w<<<g1(128*112), b256, 0, stream>>>(pr_fc_w,  h_prfc,  128, 100, 128, 112);
    pack_w<<<g1(128*112), b256, 0, stream>>>(pr_cv_w,  h_prcv,  128, 100, 128, 112);
    pack_w<<<g1(128*64),  b256, 0, stream>>>(cl_fc1_w, h_clfc1, 100, 64, 128, 64);
    pack_w<<<g1(128*64),  b256, 0, stream>>>(cl_cv1_w, h_clcv1, 100, 64, 128, 64);

    // ---- GCN normalization ----
    zero_f32<<<g1(N), b256, 0, stream>>>(dinv, N);
    deg_kernel<<<g1(E), b256, 0, stream>>>(edst, dinv, E);
    dinv_kernel<<<g1(N), b256, 0, stream>>>(dinv, N);
    norm_kernel<<<g1(E), b256, 0, stream>>>(esrc, edst, dinv, nrm, E);

    // ---- encoders: out = relu(x@fc+fcb) + cvb ; then scatter-add gcn msgs ----
    struct { const float* xs; const _Float16 *fw, *cw; const float *fb, *fbb; float* o; } enc[4] = {
        { x + 16, h_e12_fc, h_e12_cv, e12_fc_b, e12_cv_b, B[0] },   // e1  <- f1
        { x + 32, h_e12_fc, h_e12_cv, e12_fc_b, e12_cv_b, B[1] },   // e21 <- f2
        { x + 32, h_e23_fc, h_e23_cv, e23_fc_b, e23_cv_b, B[2] },   // e23 <- f2
        { x + 0,  h_e23_fc, h_e23_cv, e23_fc_b, e23_cv_b, B[3] },   // e3  <- f3
    };
    for (int i = 0; i < 4; ++i) {
        gemm(stream, enc[i].xs, 48, 16, enc[i].xs, 48, enc[i].fw,
             enc[i].fb, enc[i].fbb, 128, enc[i].o, 128, 16, 128, 2);
        gemm(stream, enc[i].xs, 48, 16, enc[i].xs, 48, enc[i].cw,
             nullptr, nullptr, 128, B[4], 128, 16, 128, 0);          // h = x@cv_w
        scatter_kernel<<<g1(E * 32), b256, 0, stream>>>(B[4], enc[i].o, esrc, edst, nrm, E, 32, 128);
    }

    // ---- mlp on e21/e23, then fc1 on concat ----
    gemm(stream, B[1], 128, 128, B[1], 128, h_mlp, mlp_b, nullptr, 128, B[4], 128, 128, 128, 1);
    gemm(stream, B[2], 128, 128, B[2], 128, h_mlp, mlp_b, nullptr, 128, B[5], 128, 128, 128, 1);
    gemm(stream, B[4], 128, 128, B[5], 128, h_fc1, fc1_b, nullptr, 128, B[1], 128, 256, 128, 0); // e2

    // ---- attention: scores then softmax-weighted fusion ----
    const float* emb[3] = { B[0], B[1], B[3] };
    for (int k = 0; k < 3; ++k) {
        gemm(stream, emb[k], 128, 128, emb[k], 128, h_att, nullptr, nullptr, 128, B[4], 128, 128, 128, 0);
        att_reduce<<<g1(N * 32), b256, 0, stream>>>(B[4], att_u, s3, N, k);
    }
    softmax_fuse<<<g1(N * 128), b256, 0, stream>>>(B[0], B[1], B[3], s3, B[2], N); // fused -> B2

    // ---- projection encoder (D_PROJ=100 padded to 112, row stride 128) ----
    gemm(stream, B[2], 128, 128, B[2], 128, h_prfc, pr_fc_b, pr_cv_b, 100, B[5], 128, 128, 112, 2);
    gemm(stream, B[2], 128, 128, B[2], 128, h_prcv, nullptr, nullptr, 100, B[4], 128, 128, 112, 0);
    scatter_kernel<<<g1(E * 28), b256, 0, stream>>>(B[4], B[5], esrc, edst, nrm, E, 28, 128);

    // ---- classifier head ----
    gemm(stream, B[5], 128, 100, B[5], 128, h_clcv1, nullptr, nullptr, 64, B[4], 64, 100, 64, 0);
    fillb<<<g1(N * 64), b256, 0, stream>>>(B[0], cl_cv1_b, N * 64, 64);          // G = bias
    scatter_kernel<<<g1(E * 16), b256, 0, stream>>>(B[4], B[0], esrc, edst, nrm, E, 16, 64);
    gemm(stream, B[5], 128, 100, B[5], 128, h_clfc1, cl_fc1_b, nullptr, 64, B[1], 64, 100, 64, 1);
    combine_relu<<<g1(N * 64), b256, 0, stream>>>(B[1], B[0], B[3], N * 64);     // h -> B3

    // ---- final: out = h@fc2 + fc2b + cv2b + scatter(norm * (h@cv2)[src]) ----
    final_init<<<g1(N), b256, 0, stream>>>(B[3], cl_fc2_w, cl_fc2_b, cl_cv2_w, cl_cv2_b, out, tbuf, N);
    scatter1<<<g1(E), b256, 0, stream>>>(tbuf, out, esrc, edst, nrm, E);
}
